// TwoPathLowRankLinear_2259152798355
// MI455X (gfx1250) — 8-bit matrix operand order repaired and checked by reading, compiled, not run on hardware
//
#include <hip/hip_runtime.h>
#include <hip/hip_fp16.h>
#include <cstdint>
#include <cstddef>

typedef __attribute__((ext_vector_type(8))) int v8i;

constexpr int IN_F   = 4096;
constexpr int OUT_F  = 4096;
constexpr int RANK   = 1024;
constexpr int R_HIGH = 256;

constexpr int LDS_ROW  = 80;
constexpr int A_TILE_B = 128 * LDS_ROW;
constexpr int B_TILE_B = 64 * LDS_ROW;
constexpr int SMEM_B   = 2 * A_TILE_B + 2 * B_TILE_B;

#define ASYNC_LOAD_B64(ldsoff, gptr)                                          \
  asm volatile("global_load_async_to_lds_b64 %0, %1, off"                     \
               :: "v"(ldsoff), "v"(gptr) : "memory")
#define ASYNC_LOAD_B128(ldsoff, gptr)                                         \
  asm volatile("global_load_async_to_lds_b128 %0, %1, off"                    \
               :: "v"(ldsoff), "v"(gptr) : "memory")
#define WAIT_ASYNC_5() asm volatile("s_wait_asynccnt 5" ::: "memory")
#define WAIT_ASYNC_0() asm volatile("s_wait_asynccnt 0" ::: "memory")

__device__ __forceinline__ float block_reduce_max_256(float v, float* red) {
  int tid = threadIdx.x;
  red[tid] = v;
  __syncthreads();
  #pragma unroll
  for (int s = 128; s > 0; s >>= 1) {
    if (tid < s) red[tid] = fmaxf(red[tid], red[tid + s]);
    __syncthreads();
  }
  float r = red[0];
  __syncthreads();
  return r;
}

__global__ __launch_bounds__(256) void quant_x_kernel(const float* __restrict__ x,
                                                      int8_t* __restrict__ xq,
                                                      float* __restrict__ xs) {
  __shared__ float red[256];
  const int row = blockIdx.x;
  const int tid = threadIdx.x;
  const float4* xr = (const float4*)(x + (size_t)row * IN_F) + tid * 4;
  float4 v[4];
  float am = 0.f;
  #pragma unroll
  for (int i = 0; i < 4; ++i) {
    v[i] = xr[i];
    am = fmaxf(am, fmaxf(fmaxf(fabsf(v[i].x), fabsf(v[i].y)),
                         fmaxf(fabsf(v[i].z), fabsf(v[i].w))));
  }
  float mx = block_reduce_max_256(am, red);
  float scale = fmaxf(mx, 1e-8f) / 127.0f;
  if (tid == 0) xs[row] = scale;
  int out[4];
  #pragma unroll
  for (int i = 0; i < 4; ++i) {
    float f[4] = {v[i].x, v[i].y, v[i].z, v[i].w};
    int packed = 0;
    #pragma unroll
    for (int c = 0; c < 4; ++c) {
      float q = rintf(f[c] / scale);
      q = fminf(fmaxf(q, -127.f), 127.f);
      packed |= ((int)q & 0xff) << (8 * c);
    }
    out[i] = packed;
  }
  ((int4*)(xq + (size_t)row * IN_F))[tid] = make_int4(out[0], out[1], out[2], out[3]);
}

__global__ __launch_bounds__(256) void quant_v_kernel(const float* __restrict__ vw,
                                                      const int* __restrict__ hi,
                                                      const int* __restrict__ lo,
                                                      int8_t* __restrict__ vq,
                                                      float* __restrict__ vsc) {
  __shared__ float red[256];
  const int j   = blockIdx.x;
  const int tid = threadIdx.x;
  const float qmax = (j < R_HIGH) ? 127.f : 7.f;
  const int src = (j < R_HIGH) ? hi[j] : lo[j - R_HIGH];
  const float4* wr = (const float4*)(vw + (size_t)src * IN_F) + tid * 4;
  float4 v[4];
  float am = 0.f;
  #pragma unroll
  for (int i = 0; i < 4; ++i) {
    v[i] = wr[i];
    am = fmaxf(am, fmaxf(fmaxf(fabsf(v[i].x), fabsf(v[i].y)),
                         fmaxf(fabsf(v[i].z), fabsf(v[i].w))));
  }
  float mx = block_reduce_max_256(am, red);
  float scale = fmaxf(mx, 1e-8f) / qmax;
  if (tid == 0) vsc[j] = __half2float(__float2half(scale));
  int out[4];
  #pragma unroll
  for (int i = 0; i < 4; ++i) {
    float f[4] = {v[i].x, v[i].y, v[i].z, v[i].w};
    int packed = 0;
    #pragma unroll
    for (int c = 0; c < 4; ++c) {
      float q = rintf(f[c] / scale);
      q = fminf(fmaxf(q, -qmax), qmax);
      packed |= ((int)q & 0xff) << (8 * c);
    }
    out[i] = packed;
  }
  ((int4*)(vq + (size_t)j * IN_F))[tid] = make_int4(out[0], out[1], out[2], out[3]);
}

__global__ __launch_bounds__(256) void quant_u_kernel(const float* __restrict__ uw,
                                                      const int* __restrict__ hi,
                                                      const int* __restrict__ lo,
                                                      int8_t* __restrict__ uq,
                                                      float* __restrict__ ush,
                                                      float* __restrict__ usl) {
  __shared__ float redH[256];
  __shared__ float redL[256];
  const int n   = blockIdx.x;
  const int tid = threadIdx.x;
  const float* ur = uw + (size_t)n * RANK;
  const bool highReg = (tid < 64);
  float vals[4];
  float am = 0.f;
  #pragma unroll
  for (int i = 0; i < 4; ++i) {
    int j = tid * 4 + i;
    int src = (j < R_HIGH) ? hi[j] : lo[j - R_HIGH];
    vals[i] = ur[src];
    am = fmaxf(am, fabsf(vals[i]));
  }
  redH[tid] = highReg ? am : 0.f;
  redL[tid] = highReg ? 0.f : am;
  __syncthreads();
  #pragma unroll
  for (int s = 128; s > 0; s >>= 1) {
    if (tid < s) {
      redH[tid] = fmaxf(redH[tid], redH[tid + s]);
      redL[tid] = fmaxf(redL[tid], redL[tid + s]);
    }
    __syncthreads();
  }
  float sh = fmaxf(redH[0], 1e-8f) / 127.f;
  float sl = fmaxf(redL[0], 1e-8f) / 7.f;
  if (tid == 0) {
    ush[n] = __half2float(__float2half(sh));
    usl[n] = __half2float(__float2half(sl));
  }
  const float scale = highReg ? sh : sl;
  const float qm    = highReg ? 127.f : 7.f;
  int packed = 0;
  #pragma unroll
  for (int c = 0; c < 4; ++c) {
    float q = rintf(vals[c] / scale);
    q = fminf(fmaxf(q, -qm), qm);
    packed |= ((int)q & 0xff) << (8 * c);
  }
  ((int*)(uq + (size_t)n * RANK))[tid] = packed;
}

__global__ __launch_bounds__(256) void quant_y1_kernel(const float* __restrict__ y1,
                                                       int8_t* __restrict__ y1q,
                                                       float* __restrict__ ysh,
                                                       float* __restrict__ ysl) {
  __shared__ float redH[256];
  __shared__ float redL[256];
  const int row = blockIdx.x;
  const int tid = threadIdx.x;
  const float4 v = ((const float4*)(y1 + (size_t)row * RANK))[tid];
  const bool highReg = (tid < 64);
  float am = fmaxf(fmaxf(fabsf(v.x), fabsf(v.y)), fmaxf(fabsf(v.z), fabsf(v.w)));
  redH[tid] = highReg ? am : 0.f;
  redL[tid] = highReg ? 0.f : am;
  __syncthreads();
  #pragma unroll
  for (int s = 128; s > 0; s >>= 1) {
    if (tid < s) {
      redH[tid] = fmaxf(redH[tid], redH[tid + s]);
      redL[tid] = fmaxf(redL[tid], redL[tid + s]);
    }
    __syncthreads();
  }
  float sh = fmaxf(redH[0], 1e-8f) / 127.f;
  float sl = fmaxf(redL[0], 1e-8f) / 127.f;
  if (tid == 0) { ysh[row] = sh; ysl[row] = sl; }
  const float scale = highReg ? sh : sl;
  float f[4] = {v.x, v.y, v.z, v.w};
  int packed = 0;
  #pragma unroll
  for (int c = 0; c < 4; ++c) {
    float q = rintf(f[c] / scale);
    q = fminf(fmaxf(q, -127.f), 127.f);
    packed |= ((int)q & 0xff) << (8 * c);
  }
  ((int*)(y1q + (size_t)row * RANK))[tid] = packed;
}

__device__ __forceinline__ void issue_tile(const int8_t* ga0, int lda,
                                           const int8_t* gb0,
                                           unsigned la0, unsigned lb0,
                                           int koff, int buf) {
  #pragma unroll
  for (int i = 0; i < 4; ++i) {
    unsigned lds = la0 + buf * A_TILE_B + i * 32 * LDS_ROW;
    const int8_t* g = ga0 + (size_t)koff + (size_t)i * 32 * lda;
    ASYNC_LOAD_B64(lds, g);
  }
  {
    unsigned lds = lb0 + buf * B_TILE_B;
    const int8_t* g = gb0 + koff;
    ASYNC_LOAD_B128(lds, g);
  }
}

__device__ __forceinline__ v8i frag8(const char* p) {
  const int4* q = (const int4*)p;
  int4 x = q[0];
  int4 y = q[1];
  v8i f;
  f[0] = x.x; f[1] = x.y; f[2] = x.z; f[3] = x.w;
  f[4] = y.x; f[5] = y.y; f[6] = y.z; f[7] = y.w;
  return f;
}

#define MMA_STEP(BUF, ACC)                                                     \
  {                                                                            \
    _Pragma("unroll")                                                          \
    for (int mt = 0; mt < 2; ++mt) {                                           \
      v8i a = frag8(&smem[(BUF) * A_TILE_B +                                   \
                          (wm * 32 + mt * 16 + lm) * LDS_ROW + hlf * 32]);     \
      _Pragma("unroll")                                                        \
      for (int nt = 0; nt < 2; ++nt) {                                         \
        v8i b = frag8(&smem[2 * A_TILE_B + (BUF) * B_TILE_B +                  \
                            (wn * 32 + nt * 16 + lm) * LDS_ROW + hlf * 32]);   \
        ACC[mt][nt] = __builtin_amdgcn_wmma_i32_16x16x64_iu8(                  \
            true, a, true, b, ACC[mt][nt], false, false);                      \
      }                                                                        \
    }                                                                          \
  }

#define SETUP_ASYNC_BASES(APTR, LDA, BPTR, LDB)                                \
  const unsigned sbase = (unsigned)(size_t)(&smem[0]);                         \
  const int arow0 = tid >> 3, ae = tid & 7;                                    \
  const int brow0 = tid >> 2, bcg = tid & 3;                                   \
  const int8_t* ga0 = (APTR) + (size_t)arow0 * (LDA) + ae * 8;                 \
  const int8_t* gb0 = (BPTR) + (size_t)brow0 * (LDB) + bcg * 16;               \
  const unsigned la0 =                                                         \
      sbase + arow0 * LDS_ROW + ((ae >> 1) & 1) * 32 + (ae >> 2) * 16 + (ae & 1) * 8; \
  const unsigned lb0 = sbase + 2 * A_TILE_B + brow0 * LDS_ROW +                \
                       (bcg & 1) * 32 + (bcg >> 1) * 16;

__global__ __launch_bounds__(256) void gemm1_kernel(const int8_t* __restrict__ xq,
                                                    const float* __restrict__ xs,
                                                    const int8_t* __restrict__ vq,
                                                    const float* __restrict__ vsc,
                                                    float* __restrict__ y1) {
  __shared__ __align__(16) char smem[SMEM_B];
  const int tid  = threadIdx.x;
  const int m0   = blockIdx.x * 128;
  const int n0   = blockIdx.y * 64;
  const int lane = tid & 31;
  const int wave = tid >> 5;
  const int wm   = wave & 3;
  const int wn   = wave >> 2;
  const int lm   = lane & 15;
  const int hlf  = lane >> 4;

  SETUP_ASYNC_BASES(xq + (size_t)m0 * IN_F, IN_F, vq + (size_t)n0 * IN_F, IN_F)

  v8i acc[2][2] = {};

  constexpr int STEPS = IN_F / 64;
  issue_tile(ga0, IN_F, gb0, la0, lb0, 0, 0);
  for (int s = 0; s < STEPS - 2; s += 2) {
    issue_tile(ga0, IN_F, gb0, la0, lb0, (s + 1) * 64, 1);
    WAIT_ASYNC_5(); __syncthreads();
    MMA_STEP(0, acc); __syncthreads();
    issue_tile(ga0, IN_F, gb0, la0, lb0, (s + 2) * 64, 0);
    WAIT_ASYNC_5(); __syncthreads();
    MMA_STEP(1, acc); __syncthreads();
  }
  issue_tile(ga0, IN_F, gb0, la0, lb0, (STEPS - 1) * 64, 1);
  WAIT_ASYNC_5(); __syncthreads();
  MMA_STEP(0, acc); __syncthreads();
  WAIT_ASYNC_0(); __syncthreads();
  MMA_STEP(1, acc);

  #pragma unroll
  for (int mt = 0; mt < 2; ++mt) {
    #pragma unroll
    for (int nt = 0; nt < 2; ++nt) {
      int col = n0 + wn * 32 + nt * 16 + lm;
      float cs = vsc[col];
      int row0 = m0 + wm * 32 + mt * 16 + hlf * 8;
      #pragma unroll
      for (int r = 0; r < 8; ++r) {
        int row = row0 + r;
        y1[(size_t)row * RANK + col] = (float)acc[mt][nt][r] * xs[row] * cs;
      }
    }
  }
}

__global__ __launch_bounds__(256) void gemm2_kernel(const int8_t* __restrict__ y1q,
                                                    const float* __restrict__ ysh,
                                                    const float* __restrict__ ysl,
                                                    const int8_t* __restrict__ uq,
                                                    const float* __restrict__ ush,
                                                    const float* __restrict__ usl,
                                                    const float* __restrict__ bias,
                                                    float* __restrict__ out) {
  __shared__ __align__(16) char smem[SMEM_B];
  const int tid  = threadIdx.x;
  const int m0   = blockIdx.x * 128;
  const int n0   = blockIdx.y * 64;
  const int lane = tid & 31;
  const int wave = tid >> 5;
  const int wm   = wave & 3;
  const int wn   = wave >> 2;
  const int lm   = lane & 15;
  const int hlf  = lane >> 4;

  SETUP_ASYNC_BASES(y1q + (size_t)m0 * RANK, RANK, uq + (size_t)n0 * RANK, RANK)

  v8i accH[2][2] = {};
  v8i accL[2][2] = {};

  issue_tile(ga0, RANK, gb0, la0, lb0, 0, 0);
  issue_tile(ga0, RANK, gb0, la0, lb0, 1 * 64, 1);
  WAIT_ASYNC_5(); __syncthreads();
  MMA_STEP(0, accH); __syncthreads();
  issue_tile(ga0, RANK, gb0, la0, lb0, 2 * 64, 0);
  WAIT_ASYNC_5(); __syncthreads();
  MMA_STEP(1, accH); __syncthreads();
  issue_tile(ga0, RANK, gb0, la0, lb0, 3 * 64, 1);
  WAIT_ASYNC_5(); __syncthreads();
  MMA_STEP(0, accH); __syncthreads();
  issue_tile(ga0, RANK, gb0, la0, lb0, 4 * 64, 0);
  WAIT_ASYNC_5(); __syncthreads();
  MMA_STEP(1, accH); __syncthreads();
  for (int s = 4; s < 14; s += 2) {
    issue_tile(ga0, RANK, gb0, la0, lb0, (s + 1) * 64, 1);
    WAIT_ASYNC_5(); __syncthreads();
    MMA_STEP(0, accL); __syncthreads();
    issue_tile(ga0, RANK, gb0, la0, lb0, (s + 2) * 64, 0);
    WAIT_ASYNC_5(); __syncthreads();
    MMA_STEP(1, accL); __syncthreads();
  }
  issue_tile(ga0, RANK, gb0, la0, lb0, 15 * 64, 1);
  WAIT_ASYNC_5(); __syncthreads();
  MMA_STEP(0, accL); __syncthreads();
  WAIT_ASYNC_0(); __syncthreads();
  MMA_STEP(1, accL);

  #pragma unroll
  for (int mt = 0; mt < 2; ++mt) {
    #pragma unroll
    for (int nt = 0; nt < 2; ++nt) {
      int col = n0 + wn * 32 + nt * 16 + lm;
      float csh = ush[col];
      float csl = usl[col];
      float b16 = __half2float(__float2half(bias[col]));
      int row0 = m0 + wm * 32 + mt * 16 + hlf * 8;
      #pragma unroll
      for (int r = 0; r < 8; ++r) {
        int row = row0 + r;
        out[(size_t)row * OUT_F + col] =
            (float)accH[mt][nt][r] * ysh[row] * csh +
            (float)accL[mt][nt][r] * ysl[row] * csl + b16;
      }
    }
  }
}

extern "C" void kernel_launch(void* const* d_in, const int* in_sizes, int n_in,
                              void* d_out, int out_size, void* d_ws, size_t ws_size,
                              hipStream_t stream) {
  const float* x    = (const float*)d_in[0];
  const float* uw   = (const float*)d_in[1];
  const float* vw   = (const float*)d_in[2];
  const float* bias = (const float*)d_in[3];
  const int*   hi   = (const int*)d_in[4];
  const int*   lo   = (const int*)d_in[5];
  const int M = in_sizes[0] / IN_F;

  size_t off = 0;
  auto carve = [&](size_t bytes) -> void* {
    void* p = (char*)d_ws + off;
    off += (bytes + 255) & ~(size_t)255;
    return p;
  };
  int8_t* xq  = (int8_t*)carve((size_t)M * IN_F);
  float*  xs  = (float*)carve((size_t)M * sizeof(float));
  int8_t* vq  = (int8_t*)carve((size_t)RANK * IN_F);
  float*  vsc = (float*)carve((size_t)RANK * sizeof(float));
  int8_t* uq  = (int8_t*)carve((size_t)OUT_F * RANK);
  float*  ush = (float*)carve((size_t)OUT_F * sizeof(float));
  float*  usl = (float*)carve((size_t)OUT_F * sizeof(float));
  float*  y1  = (float*)carve((size_t)M * RANK * sizeof(float));
  int8_t* y1q = (int8_t*)carve((size_t)M * RANK);
  float*  ysh = (float*)carve((size_t)M * sizeof(float));
  float*  ysl = (float*)carve((size_t)M * sizeof(float));
  (void)ws_size; (void)n_in; (void)out_size;

  quant_x_kernel<<<M, 256, 0, stream>>>(x, xq, xs);
  quant_v_kernel<<<RANK, 256, 0, stream>>>(vw, hi, lo, vq, vsc);
  quant_u_kernel<<<OUT_F, 256, 0, stream>>>(uw, hi, lo, uq, ush, usl);

  dim3 g1(M / 128, RANK / 64);
  gemm1_kernel<<<g1, 256, 0, stream>>>(xq, xs, vq, vsc, y1);

  quant_y1_kernel<<<M, 256, 0, stream>>>(y1, y1q, ysh, ysl);

  dim3 g2(M / 128, OUT_F / 64);
  gemm2_kernel<<<g2, 256, 0, stream>>>(y1q, ysh, ysl, uq, ush, usl, bias,
                                       (float*)d_out);
}
